// ScaledDotProductAttention_53042846105919
// MI455X (gfx1250) — compile-verified, matrix operand order checked by reading, not run on hardware
//
#include <hip/hip_runtime.h>
#include <math.h>

typedef __attribute__((ext_vector_type(16))) _Float16 v16h;
typedef __attribute__((ext_vector_type(8)))  float    v8f;
typedef __attribute__((ext_vector_type(4)))  float    v4f;

#define S_LEN 1024
#define D_DIM 64
#define NEG_VAL    (-1.0e9f)
#define NEG_THRESH (-1.0e8f)

// Load 8+8 f32 -> f16 fragment row: elements 0..7 <- p[0..7], 8..15 <- p[16..23]
// (matches CDNA5 16-bit A-matrix 16x32 per-lane layout, kb selects low/high K half)
__device__ __forceinline__ v16h load_row_f32_to_f16(const float* __restrict__ p) {
  const v4f* q = (const v4f*)p;
  v4f x0 = q[0], x1 = q[1], x2 = q[4], x3 = q[5];
  v16h r;
  r[0]=(_Float16)x0.x;  r[1]=(_Float16)x0.y;  r[2]=(_Float16)x0.z;  r[3]=(_Float16)x0.w;
  r[4]=(_Float16)x1.x;  r[5]=(_Float16)x1.y;  r[6]=(_Float16)x1.z;  r[7]=(_Float16)x1.w;
  r[8]=(_Float16)x2.x;  r[9]=(_Float16)x2.y;  r[10]=(_Float16)x2.z; r[11]=(_Float16)x2.w;
  r[12]=(_Float16)x3.x; r[13]=(_Float16)x3.y; r[14]=(_Float16)x3.z; r[15]=(_Float16)x3.w;
  return r;
}

__global__ __launch_bounds__(256, 1)
void attn_fused_kernel(const float* __restrict__ query,
                       const float* __restrict__ key,
                       const float* __restrict__ value,
                       const int*   __restrict__ mask,
                       float* __restrict__ out,     // [B,H,S,D]
                       float* __restrict__ attn) {  // [B,H,S,S]
  __shared__ float sS[16 * S_LEN];  // 64 KB: scores -> exp -> normalized P; head reused for partials

  const int qt = blockIdx.x;        // 0..63 : 16-row query tile
  const int bh = blockIdx.y;        // 0..127
  const int b  = bh >> 4;           // H == 16

  const int tid  = threadIdx.x;
  const int wave = tid >> 5;
  const int lane = tid & 31;
  const int ll   = lane & 15;       // row (A) / col (B,C)
  const int hi   = lane >> 4;       // which K/M half this lane holds
  const int kb   = hi * 8;

  const float* qp = query + ((size_t)bh * S_LEN + qt * 16) * D_DIM;
  const float* kp = key   + (size_t)bh * S_LEN * D_DIM;
  const float* vp = value + (size_t)bh * S_LEN * D_DIM;
  const int*   mp = mask  + ((size_t)b  * S_LEN + qt * 16) * S_LEN;
  float*       op = out   + ((size_t)bh * S_LEN + qt * 16) * D_DIM;
  float*       ap = attn  + ((size_t)bh * S_LEN + qt * 16) * S_LEN;

  // ---------------- Phase 1: scores = mask(Q K^T / 8) -> LDS ----------------
  v16h a0 = load_row_f32_to_f16(qp + ll * D_DIM + kb);        // K(d) 0..31
  v16h a1 = load_row_f32_to_f16(qp + ll * D_DIM + kb + 32);   // K(d) 32..63

  for (int j = wave; j < 64; j += 8) {
    if (j + 8 < 64)
      __builtin_prefetch(kp + (size_t)((j + 8) * 16 + ll) * D_DIM, 0, 1);
    const float* krow = kp + (size_t)(j * 16 + ll) * D_DIM;   // B^T row == K row
    v16h b0 = load_row_f32_to_f16(krow + kb);
    v16h b1 = load_row_f32_to_f16(krow + kb + 32);
    v8f c = {};
    c = __builtin_amdgcn_wmma_f32_16x16x32_f16(false, a0, false, b0, (short)0, c, false, false);
    c = __builtin_amdgcn_wmma_f32_16x16x32_f16(false, a1, false, b1, (short)0, c, false, false);
    const int col = j * 16 + ll;
#pragma unroll
    for (int v = 0; v < 8; ++v) {
      const int m = v + 8 * hi;                // C/D layout: lanes 0-15 -> M=v, 16-31 -> M=v+8
      float s = c[v] * 0.125f;                 // 1/sqrt(64)
      if (mp[m * S_LEN + col] == 0) s = NEG_VAL;
      sS[m * S_LEN + col] = s;
    }
  }
  __syncthreads();

  // ---------------- Phase 2: row softmax + attention write ------------------
  const int row = tid >> 4;                    // 0..15
  const int lr  = tid & 15;
  float* srow = sS + row * S_LEN;

  float mx = -INFINITY;
  for (int c = lr; c < S_LEN; c += 16) mx = fmaxf(mx, srow[c]);
#pragma unroll
  for (int off = 8; off >= 1; off >>= 1) mx = fmaxf(mx, __shfl_xor(mx, off, 16));

  float sum = 0.f;
  for (int c = lr; c < S_LEN; c += 16) {
    float s = srow[c];
    float e = (s > NEG_THRESH) ? __expf(s - mx) : 0.f;  // masked -> exactly 0
    srow[c] = e;
    sum += e;
  }
#pragma unroll
  for (int off = 8; off >= 1; off >>= 1) sum += __shfl_xor(sum, off, 16);
  const float inv = (sum > 0.f) ? (1.f / sum) : 0.f;    // all-masked row -> zeros
  __syncthreads();

  {
    v4f* s4 = (v4f*)srow;                                // 256 float4 per row
    v4f* g4 = (v4f*)(ap + (size_t)row * S_LEN);          // 16x1024 tile is contiguous in global
#pragma unroll 4
    for (int it = 0; it < 16; ++it) {
      const int i4 = it * 16 + lr;                       // coalesced across lr
      v4f p = s4[i4];
      p *= inv;
      s4[i4] = p;                                        // normalized P for phase 3
      __builtin_nontemporal_store(p, &g4[i4]);           // stream 537 MB total, don't cache
    }
  }
  __syncthreads();

  // ---------------- Phase 3: out = P @ V ------------------------------------
  const int nt = wave & 3;          // d-tile (4 x 16 = 64)
  const int kh = wave >> 2;         // K half: [0,512) or [512,1024)
  const int d0 = nt * 16;
  v8f acc = {};
  for (int kt = 0; kt < 16; ++kt) {
    const int k0 = kh * 512 + kt * 32;
    v16h pa, pb;
    const float* prow = sS + ll * S_LEN + k0 + kb;       // A: P rows from LDS
#pragma unroll
    for (int i = 0; i < 8; ++i) {
      pa[i]     = (_Float16)prow[i];
      pa[8 + i] = (_Float16)prow[16 + i];
    }
    const float* vcol = vp + (size_t)(k0 + kb) * D_DIM + d0 + ll;  // B: V column (L2-hot)
#pragma unroll
    for (int i = 0; i < 8; ++i) {
      pb[i]     = (_Float16)vcol[i * D_DIM];
      pb[8 + i] = (_Float16)vcol[(16 + i) * D_DIM];
    }
    acc = __builtin_amdgcn_wmma_f32_16x16x32_f16(false, pa, false, pb, (short)0, acc, false, false);
  }
  __syncthreads();                   // everyone done reading sS; safe to reuse as scratch
  if (kh == 1) {
#pragma unroll
    for (int v = 0; v < 8; ++v)
      sS[nt * 256 + (v + 8 * hi) * 16 + ll] = acc[v];
  }
  __syncthreads();
  if (kh == 0) {
#pragma unroll
    for (int v = 0; v < 8; ++v) {
      const int m = v + 8 * hi;
      op[m * D_DIM + d0 + ll] = acc[v] + sS[nt * 256 + m * 16 + ll];
    }
  }
}

extern "C" void kernel_launch(void* const* d_in, const int* in_sizes, int n_in,
                              void* d_out, int out_size, void* d_ws, size_t ws_size,
                              hipStream_t stream) {
  (void)in_sizes; (void)n_in; (void)out_size; (void)d_ws; (void)ws_size;
  const float* q = (const float*)d_in[0];
  const float* k = (const float*)d_in[1];
  const float* v = (const float*)d_in[2];
  const int*   m = (const int*)d_in[3];
  float* out  = (float*)d_out;
  const size_t OUT_ELEMS = (size_t)8 * 16 * 1024 * 64;   // B*H*S*D
  float* attn = out + OUT_ELEMS;                          // tuple: (out, attention) concatenated

  dim3 grid(1024 / 16, 8 * 16);   // (q-tiles, B*H) = (64, 128)
  attn_fused_kernel<<<grid, 256, 0, stream>>>(q, k, v, m, out, attn);
}
